// EncoderLayer_19473381720284
// MI455X (gfx1250) — matrix operand order repaired and checked by reading, not compiled, not run on hardware
//
#include <hip/hip_runtime.h>
#include <hip/hip_bf16.h>
#include <math.h>

// ---------------------------------------------------------------------------
// Types for CDNA5 WMMA (wave32): v_wmma_f32_16x16x32_bf16
// ---------------------------------------------------------------------------
typedef __bf16 bf16_t;
typedef __attribute__((ext_vector_type(16))) __bf16 v16bf;
typedef __attribute__((ext_vector_type(8)))  __bf16 v8bf;
typedef __attribute__((ext_vector_type(8)))  float  v8f;
typedef __attribute__((ext_vector_type(4)))  unsigned int u32x4;
typedef __attribute__((ext_vector_type(4)))  int i32x4;
typedef __attribute__((ext_vector_type(8)))  int i32x8;

#define D_MODEL 768
#define NHEADS  12
#define HDIM    64
#define HIDDEN  3072
#define SEQ     2048
#define BATCH   2
#define ROWS    (BATCH * SEQ)   // 4096

#ifndef __has_builtin
#define __has_builtin(x) 0
#endif

#if defined(__HIP_DEVICE_COMPILE__) && __has_builtin(__builtin_amdgcn_tensor_load_to_lds)
#define HAVE_TDM 1
#else
#define HAVE_TDM 0
#endif

__device__ __forceinline__ bf16_t f2bf(float f) {
  union { float f; unsigned u; } c; c.f = f;
  unsigned u = c.u;
  u += 0x7fffu + ((u >> 16) & 1u);          // round-to-nearest-even
  unsigned short h = (unsigned short)(u >> 16);
  union { unsigned short s; bf16_t b; } r; r.s = h;
  return r.b;
}

union frag16 { v16bf v; v8bf h[2]; };

__device__ __forceinline__ v16bf load_frag2(const bf16_t* p0, const bf16_t* p1) {
  frag16 f;
  f.h[0] = *(const v8bf*)p0;
  f.h[1] = *(const v8bf*)p1;
  return f.v;
}

__device__ __forceinline__ v8f wmma_bf16(v16bf a, v16bf b, v8f c) {
  // (neg_a, A, neg_b, B, c_mod, C, reuse_a, reuse_b)
  return __builtin_amdgcn_wmma_f32_16x16x32_bf16(false, a, false, b, (short)0, c,
                                                 false, false);
}

__device__ __forceinline__ void wait_tensorcnt0() {
#if __has_builtin(__builtin_amdgcn_s_wait_tensorcnt)
  __builtin_amdgcn_s_wait_tensorcnt(0);
#else
  asm volatile("s_wait_tensorcnt 0x0" ::: "memory");
#endif
}

#if HAVE_TDM
// ---------------------------------------------------------------------------
// Tensor Data Mover: 2D tile (bf16) global -> LDS with row padding.
// D# layout per CDNA5 ISA:
//  group0: [1:0]=count, [63:32]=lds_addr, [120:64]=global_addr, [127:126]=type(2)
//  group1: [17:16]=data_size(1=2B), [20]=pad_en, [24:22]=pad_interval,
//          [31:25]=pad_amount, [79:48]=tensor_dim0, [111:80]=tensor_dim1,
//          [127:112]=tile_dim0, [143:128]=tile_dim1, [207:160]=dim0_stride
// ---------------------------------------------------------------------------
__device__ __forceinline__ void tdm_load_2d(unsigned lds_off, const void* gptr,
                                            unsigned tensor_d0, unsigned tensor_d1,
                                            unsigned tile_d0, unsigned tile_d1,
                                            unsigned stride0_elems,
                                            unsigned pad_interval_code,
                                            unsigned pad_amount_code) {
  unsigned long long ga = (unsigned long long)(uintptr_t)gptr;
  u32x4 g0;
  g0[0] = 1u;                                   // count=1, user descriptor
  g0[1] = lds_off;                              // LDS byte address
  g0[2] = (unsigned)(ga & 0xFFFFFFFFu);
  g0[3] = ((unsigned)((ga >> 32) & 0x1FFFFFFu)) | (2u << 30);  // type=2 (image)
  i32x8 g1;
  unsigned d0 = (1u << 16)                      // data_size = 2 bytes
              | (1u << 20)                      // pad_enable
              | (pad_interval_code << 22)
              | (pad_amount_code << 25);
  g1[0] = (int)d0;
  g1[1] = (int)((tensor_d0 & 0xFFFFu) << 16);   // [15:0] atomic_barrier_addr = 0
  g1[2] = (int)((tensor_d0 >> 16) | ((tensor_d1 & 0xFFFFu) << 16));
  g1[3] = (int)((tensor_d1 >> 16) | (tile_d0 << 16));
  g1[4] = (int)tile_d1;                         // tile_dim2 = 0
  g1[5] = (int)stride0_elems;                   // tensor_dim0_stride[31:0]
  g1[6] = 0;
  g1[7] = 0;
  i32x4 z4 = {0, 0, 0, 0};
#if defined(__clang_major__) && __clang_major__ >= 23
  i32x8 z8 = {0, 0, 0, 0, 0, 0, 0, 0};
  __builtin_amdgcn_tensor_load_to_lds(g0, g1, z4, z4, z8, 0);
#else
  __builtin_amdgcn_tensor_load_to_lds(g0, g1, z4, z4, 0);
#endif
}
#endif  // HAVE_TDM

// ---------------------------------------------------------------------------
// fp32 -> bf16 weight conversion (grid-stride)
// ---------------------------------------------------------------------------
__global__ void cvt_f32_bf16(const float* __restrict__ in, bf16_t* __restrict__ out,
                             int n) {
  int i = blockIdx.x * blockDim.x + threadIdx.x;
  int stride = gridDim.x * blockDim.x;
  for (; i < n; i += stride) out[i] = f2bf(in[i]);
}

// ---------------------------------------------------------------------------
// LayerNorm over D_MODEL=768, one row per block (256 threads, 3 elems/thread)
// ---------------------------------------------------------------------------
__device__ __forceinline__ float block_reduce_sum(float s, float* red) {
  int lane = threadIdx.x & 31, wave = threadIdx.x >> 5;
  #pragma unroll
  for (int off = 16; off; off >>= 1) s += __shfl_xor(s, off, 32);
  if (lane == 0) red[wave] = s;
  __syncthreads();
  float t = (threadIdx.x < 8) ? red[threadIdx.x] : 0.0f;
  if (wave == 0) {
    #pragma unroll
    for (int off = 4; off; off >>= 1) t += __shfl_xor(t, off, 8);
    if (lane == 0) red[0] = t;
  }
  __syncthreads();
  float r = red[0];
  __syncthreads();
  return r;
}

__global__ __launch_bounds__(256) void layernorm_bf16_kernel(
    const float* __restrict__ x, const float* __restrict__ g,
    const float* __restrict__ b, bf16_t* __restrict__ out) {
  __shared__ float red[8];
  int row = blockIdx.x;
  const float* xr = x + (size_t)row * D_MODEL;
  int t = threadIdx.x;
  float v0 = xr[t], v1 = xr[t + 256], v2 = xr[t + 512];
  float mean = block_reduce_sum(v0 + v1 + v2, red) * (1.0f / D_MODEL);
  float d0 = v0 - mean, d1 = v1 - mean, d2 = v2 - mean;
  float var = block_reduce_sum(d0 * d0 + d1 * d1 + d2 * d2, red) * (1.0f / D_MODEL);
  float rstd = rsqrtf(var + 1e-6f);
  bf16_t* orow = out + (size_t)row * D_MODEL;
  orow[t]       = f2bf(d0 * rstd * g[t]       + b[t]);
  orow[t + 256] = f2bf(d1 * rstd * g[t + 256] + b[t + 256]);
  orow[t + 512] = f2bf(d2 * rstd * g[t + 512] + b[t + 512]);
}

// ---------------------------------------------------------------------------
// Tiled bf16 WMMA GEMM: C[M,N] = A[M,K] * B[K,N], A/B row-major bf16.
// Block 256 thr = 8 waves; block tile 128x128; wave tile 32(M)x64(N); K-step 32.
// A tile staged by the Tensor Data Mover (row pad 16 DWORDs + 4 -> LDA=40);
// B tile staged transposed by the threads so B-fragments are contiguous loads.
// MODE 0: QKV split -> Q/K/V bf16 buffers  [bh][n][d]
// MODE 1: out_f32 = residual + bias + acc          (attn proj)
// MODE 2: out_bf16 = gelu_exact(acc + bias)        (FC1)
// MODE 3: out_f32 = residual + bias + acc          (FC2 -> d_out)
// ---------------------------------------------------------------------------
#define BM 128
#define BN 128
#define BK 32
#define LDA 40
#define LDB 40

template <int MODE>
__global__ __launch_bounds__(256) void gemm_bf16_wmma(
    const bf16_t* __restrict__ A, const bf16_t* __restrict__ B,
    int M, int N, int K,
    const float* __restrict__ bias, const float* __restrict__ residual,
    float* __restrict__ outf, bf16_t* __restrict__ outq,
    bf16_t* __restrict__ outk, bf16_t* __restrict__ outv) {
  __shared__ bf16_t As[BM * LDA];
  __shared__ bf16_t Bt[BN * LDB];

  int tid = threadIdx.x;
  int lane = tid & 31, wave = tid >> 5;
  int wm = wave & 3;        // 0..3 along M
  int wn = wave >> 2;       // 0..1 along N
  int halfsel = lane >> 4;  // 0|1
  int l16 = lane & 15;
  int bm0 = blockIdx.y * BM;
  int bn0 = blockIdx.x * BN;

  v8f zacc = {};
  v8f acc[2][4];
  #pragma unroll
  for (int i = 0; i < 2; i++)
    #pragma unroll
    for (int j = 0; j < 4; j++) acc[i][j] = zacc;

  for (int kc = 0; kc < K; kc += BK) {
#if HAVE_TDM
    // stage A: 128x32 bf16 tile via TDM; rows of 16 DWORDs padded by 4 DWORDs
    if (wave == 0) {
      tdm_load_2d((unsigned)(uintptr_t)&As[0],
                  A + (size_t)bm0 * K + kc,
                  (unsigned)K, (unsigned)M, BK, BM, (unsigned)K,
                  /*interval 16 dw*/ 3u, /*pad 4 dw*/ 3u);
    }
#else
    {
      int row = tid >> 1, col = (tid & 1) * 16;
      const bf16_t* src = A + (size_t)(bm0 + row) * K + kc + col;
      bf16_t* dst = &As[row * LDA + col];
      *(v8bf*)dst       = *(const v8bf*)src;
      *(v8bf*)(dst + 8) = *(const v8bf*)(src + 8);
    }
#endif
    // stage B transposed: Bt[n][k]
    {
      int kr = tid >> 3;            // 0..31
      int nc = (tid & 7) * 16;      // 0..112
      const bf16_t* src = B + (size_t)(kc + kr) * N + bn0 + nc;
      v8bf b0 = *(const v8bf*)src;
      v8bf b1 = *(const v8bf*)(src + 8);
      #pragma unroll
      for (int j = 0; j < 8; j++) Bt[(nc + j) * LDB + kr] = b0[j];
      #pragma unroll
      for (int j = 0; j < 8; j++) Bt[(nc + 8 + j) * LDB + kr] = b1[j];
      if (kc + BK < K) {
        __builtin_prefetch(B + (size_t)(kc + BK + kr) * N + bn0 + nc, 0, 1);
        __builtin_prefetch(A + (size_t)(bm0 + (tid >> 1)) * K + kc + BK, 0, 1);
      }
    }
#if HAVE_TDM
    if (wave == 0) wait_tensorcnt0();
#endif
    __syncthreads();

    v16bf af[2];
    #pragma unroll
    for (int mt = 0; mt < 2; mt++) {
      int row = wm * 32 + mt * 16 + l16;
      int kb = halfsel * 8;  // A frag: half0 K 0..7 & 16..23, half1 K 8..15 & 24..31
      af[mt] = load_frag2(&As[row * LDA + kb], &As[row * LDA + kb + 16]);
    }
    #pragma unroll
    for (int nt = 0; nt < 4; nt++) {
      int n = wn * 64 + nt * 16 + l16;
      int koff = halfsel * 8;  // B frag: half0 K 0..7 & 16..23, half1 K 8..15 & 24..31
      v16bf bfr = load_frag2(&Bt[n * LDB + koff], &Bt[n * LDB + koff + 16]);
      #pragma unroll
      for (int mt = 0; mt < 2; mt++) acc[mt][nt] = wmma_bf16(af[mt], bfr, acc[mt][nt]);
    }
    __syncthreads();
  }

  // epilogue: C layout — VGPR i holds row (i + 8*halfsel), col l16
  #pragma unroll
  for (int mt = 0; mt < 2; mt++) {
    #pragma unroll
    for (int nt = 0; nt < 4; nt++) {
      int gn = bn0 + wn * 64 + nt * 16 + l16;
      #pragma unroll
      for (int i = 0; i < 8; i++) {
        int gm = bm0 + wm * 32 + mt * 16 + i + halfsel * 8;
        float v = acc[mt][nt][i];
        if (MODE == 0) {
          int which = gn / D_MODEL;
          int rem = gn - which * D_MODEL;
          int head = rem >> 6, d = rem & 63;
          int bb = gm >> 11, nn = gm & (SEQ - 1);
          size_t idx = ((size_t)(bb * NHEADS + head) * SEQ + nn) * HDIM + d;
          bf16_t* dst = (which == 0) ? outq : (which == 1) ? outk : outv;
          dst[idx] = f2bf(v);
        } else if (MODE == 1 || MODE == 3) {
          size_t idx = (size_t)gm * N + gn;
          outf[idx] = residual[idx] + bias[gn] + v;
        } else {  // MODE 2: exact GELU
          float tt = v + bias[gn];
          float gl = 0.5f * tt * (1.0f + erff(tt * 0.70710678118654752f));
          outq[(size_t)gm * N + gn] = f2bf(gl);
        }
      }
    }
  }
}

// ---------------------------------------------------------------------------
// Flash attention with WMMA. Grid: (SEQ/64, BATCH*NHEADS). Block 128 thr = 4 waves.
// K tile staged by TDM (rows 32 DWORDs + 4 pad -> ALD=72); V staged transposed
// by the threads; P staged to LDS bf16; online softmax in fp32.
// ---------------------------------------------------------------------------
#define ALD 72

__global__ __launch_bounds__(128) void flash_attn_wmma(
    const bf16_t* __restrict__ Qb, const bf16_t* __restrict__ Kb,
    const bf16_t* __restrict__ Vb, bf16_t* __restrict__ ctx) {
  __shared__ bf16_t Ks[64 * ALD];  // [key][d]
  __shared__ bf16_t Vt[64 * ALD];  // [d][key]  (transposed for B-fragments)
  __shared__ bf16_t Ps[64 * ALD];  // [qrow][key]

  int tid = threadIdx.x;
  int lane = tid & 31, wave = tid >> 5;
  int halfsel = lane >> 4, l16 = lane & 15;
  int bh = blockIdx.y;
  int q0 = blockIdx.x * 64;
  const bf16_t* Qh = Qb + (size_t)bh * SEQ * HDIM;
  const bf16_t* Kh = Kb + (size_t)bh * SEQ * HDIM;
  const bf16_t* Vh = Vb + (size_t)bh * SEQ * HDIM;

  // Q fragments for this wave's 16 rows (2 contraction chunks of 32)
  v16bf qf[2];
  #pragma unroll
  for (int kc = 0; kc < 2; kc++) {
    int row = q0 + wave * 16 + l16;
    int kb = kc * 32 + halfsel * 8;
    const bf16_t* qp = Qh + (size_t)row * HDIM + kb;
    qf[kc] = load_frag2(qp, qp + 16);
  }

  v8f zacc = {};
  v8f oacc[4];
  #pragma unroll
  for (int j = 0; j < 4; j++) oacc[j] = zacc;
  float rmax[8], rsum[8];
  #pragma unroll
  for (int i = 0; i < 8; i++) { rmax[i] = -1e30f; rsum[i] = 0.0f; }

  for (int kt = 0; kt < SEQ; kt += 64) {
#if HAVE_TDM
    // stage K tile 64x64 bf16 via TDM (row 32 DWORDs + 4 DWORDs pad -> ALD=72)
    if (wave == 0) {
      tdm_load_2d((unsigned)(uintptr_t)&Ks[0],
                  Kh + (size_t)kt * HDIM,
                  HDIM, SEQ, HDIM, 64, HDIM,
                  /*interval 32 dw*/ 4u, /*pad 4 dw*/ 3u);
    }
#endif
    // stage V tile transposed (and K manually if no TDM)
    {
      int row = tid >> 1;          // 0..63 (key index)
      int col = (tid & 1) * 32;    // 0|32 (d)
#if !HAVE_TDM
      const bf16_t* ksrc = Kh + (size_t)(kt + row) * HDIM + col;
      v8bf k0 = *(const v8bf*)ksrc;
      v8bf k1 = *(const v8bf*)(ksrc + 8);
      v8bf k2 = *(const v8bf*)(ksrc + 16);
      v8bf k3 = *(const v8bf*)(ksrc + 24);
      bf16_t* kd = &Ks[row * ALD + col];
      *(v8bf*)kd        = k0;
      *(v8bf*)(kd + 8)  = k1;
      *(v8bf*)(kd + 16) = k2;
      *(v8bf*)(kd + 24) = k3;
#endif
      const bf16_t* vsrc = Vh + (size_t)(kt + row) * HDIM + col;
      v8bf w0 = *(const v8bf*)vsrc;
      v8bf w1 = *(const v8bf*)(vsrc + 8);
      v8bf w2 = *(const v8bf*)(vsrc + 16);
      v8bf w3 = *(const v8bf*)(vsrc + 24);
      #pragma unroll
      for (int j = 0; j < 8; j++) {
        Vt[(col + j) * ALD + row]      = w0[j];
        Vt[(col + 8 + j) * ALD + row]  = w1[j];
        Vt[(col + 16 + j) * ALD + row] = w2[j];
        Vt[(col + 24 + j) * ALD + row] = w3[j];
      }
      if (kt + 64 < SEQ) {
        __builtin_prefetch(Kh + (size_t)(kt + 64 + row) * HDIM + col, 0, 1);
        __builtin_prefetch(Vh + (size_t)(kt + 64 + row) * HDIM + col, 0, 1);
      }
    }
#if HAVE_TDM
    if (wave == 0) wait_tensorcnt0();
#endif
    __syncthreads();

    // S = Q K^T * scale
    v8f sacc[4];
    #pragma unroll
    for (int nt = 0; nt < 4; nt++) {
      sacc[nt] = zacc;
      #pragma unroll
      for (int kc = 0; kc < 2; kc++) {
        int key = nt * 16 + l16;
        int koff = kc * 32 + halfsel * 8;
        v16bf kf = load_frag2(&Ks[key * ALD + koff], &Ks[key * ALD + koff + 16]);
        sacc[nt] = wmma_bf16(qf[kc], kf, sacc[nt]);
      }
    }

    // online softmax: reduce across the 16 lanes sharing each row
    float p[4][8], nmax[8], scl[8];
    #pragma unroll
    for (int i = 0; i < 8; i++) {
      float m = -1e30f;
      #pragma unroll
      for (int nt = 0; nt < 4; nt++) m = fmaxf(m, sacc[nt][i] * 0.125f);
      #pragma unroll
      for (int off = 8; off; off >>= 1) m = fmaxf(m, __shfl_xor(m, off, 16));
      float nm = fmaxf(rmax[i], m);
      scl[i] = __expf(rmax[i] - nm);
      nmax[i] = nm;
    }
    #pragma unroll
    for (int i = 0; i < 8; i++) {
      float ps = 0.0f;
      #pragma unroll
      for (int nt = 0; nt < 4; nt++) {
        float e = __expf(sacc[nt][i] * 0.125f - nmax[i]);
        p[nt][i] = e;
        ps += e;
      }
      #pragma unroll
      for (int off = 8; off; off >>= 1) ps += __shfl_xor(ps, off, 16);
      rsum[i] = rsum[i] * scl[i] + ps;
      rmax[i] = nmax[i];
    }
    #pragma unroll
    for (int j = 0; j < 4; j++)
      #pragma unroll
      for (int i = 0; i < 8; i++) oacc[j][i] *= scl[i];

    // P -> LDS (bf16) in [qrow][key] layout
    #pragma unroll
    for (int nt = 0; nt < 4; nt++) {
      #pragma unroll
      for (int i = 0; i < 8; i++) {
        int pr = wave * 16 + i + halfsel * 8;
        int pc = nt * 16 + l16;
        Ps[pr * ALD + pc] = f2bf(p[nt][i]);
      }
    }
    __syncthreads();

    // O += P V : contraction over 64 keys (2 chunks of 32)
    #pragma unroll
    for (int kc = 0; kc < 2; kc++) {
      int prow = wave * 16 + l16;
      int kb = kc * 32 + halfsel * 8;
      v16bf pf = load_frag2(&Ps[prow * ALD + kb], &Ps[prow * ALD + kb + 16]);
      #pragma unroll
      for (int nto = 0; nto < 4; nto++) {
        int d = nto * 16 + l16;
        int koff = kc * 32 + halfsel * 8;
        v16bf vf = load_frag2(&Vt[d * ALD + koff], &Vt[d * ALD + koff + 16]);
        oacc[nto] = wmma_bf16(pf, vf, oacc[nto]);
      }
    }
    __syncthreads();
  }

  // normalize + write context in [B*N, D] layout (col = h*64 + d)
  int b = bh / NHEADS, h = bh % NHEADS;
  #pragma unroll
  for (int nt = 0; nt < 4; nt++) {
    #pragma unroll
    for (int i = 0; i < 8; i++) {
      int q = q0 + wave * 16 + i + halfsel * 8;
      int col = h * HDIM + nt * 16 + l16;
      size_t row = (size_t)b * SEQ + q;
      ctx[row * D_MODEL + col] = f2bf(oacc[nt][i] / rsum[i]);
    }
  }
}

// ---------------------------------------------------------------------------
// Host-side launcher
// ---------------------------------------------------------------------------
extern "C" void kernel_launch(void* const* d_in, const int* in_sizes, int n_in,
                              void* d_out, int out_size, void* d_ws, size_t ws_size,
                              hipStream_t stream) {
  (void)in_sizes; (void)n_in; (void)out_size; (void)ws_size;

  const float* x      = (const float*)d_in[0];
  const float* ln1_g  = (const float*)d_in[1];
  const float* ln1_b  = (const float*)d_in[2];
  const float* ln2_g  = (const float*)d_in[3];
  const float* ln2_b  = (const float*)d_in[4];
  const float* w_qkv  = (const float*)d_in[5];
  const float* w_proj = (const float*)d_in[6];
  const float* b_proj = (const float*)d_in[7];
  const float* w_fc1  = (const float*)d_in[8];
  const float* b_fc1  = (const float*)d_in[9];
  const float* w_fc2  = (const float*)d_in[10];
  const float* b_fc2  = (const float*)d_in[11];
  float* out = (float*)d_out;

  char* p = (char*)d_ws;
  auto alloc = [&](size_t bytes) -> char* {
    char* r = p;
    p += (bytes + 255) & ~(size_t)255;
    return r;
  };
  bf16_t* wqkv_b  = (bf16_t*)alloc((size_t)D_MODEL * 3 * D_MODEL * 2);
  bf16_t* wproj_b = (bf16_t*)alloc((size_t)D_MODEL * D_MODEL * 2);
  bf16_t* wfc1_b  = (bf16_t*)alloc((size_t)D_MODEL * HIDDEN * 2);
  bf16_t* wfc2_b  = (bf16_t*)alloc((size_t)HIDDEN * D_MODEL * 2);
  bf16_t* xn1     = (bf16_t*)alloc((size_t)ROWS * D_MODEL * 2);
  bf16_t* Qb      = (bf16_t*)alloc((size_t)BATCH * NHEADS * SEQ * HDIM * 2);
  bf16_t* Kb      = (bf16_t*)alloc((size_t)BATCH * NHEADS * SEQ * HDIM * 2);
  bf16_t* Vb      = (bf16_t*)alloc((size_t)BATCH * NHEADS * SEQ * HDIM * 2);
  bf16_t* ctx     = (bf16_t*)alloc((size_t)ROWS * D_MODEL * 2);
  float*  x1      = (float*)alloc((size_t)ROWS * D_MODEL * 4);
  bf16_t* xn2     = (bf16_t*)alloc((size_t)ROWS * D_MODEL * 2);
  bf16_t* hbuf    = (bf16_t*)alloc((size_t)ROWS * HIDDEN * 2);

  // 1) weights fp32 -> bf16
  {
    int n;
    n = D_MODEL * 3 * D_MODEL;
    cvt_f32_bf16<<<(n + 255) / 256, 256, 0, stream>>>(w_qkv, wqkv_b, n);
    n = D_MODEL * D_MODEL;
    cvt_f32_bf16<<<(n + 255) / 256, 256, 0, stream>>>(w_proj, wproj_b, n);
    n = D_MODEL * HIDDEN;
    cvt_f32_bf16<<<(n + 255) / 256, 256, 0, stream>>>(w_fc1, wfc1_b, n);
    n = HIDDEN * D_MODEL;
    cvt_f32_bf16<<<(n + 255) / 256, 256, 0, stream>>>(w_fc2, wfc2_b, n);
  }

  // 2) LN1
  layernorm_bf16_kernel<<<ROWS, 256, 0, stream>>>(x, ln1_g, ln1_b, xn1);

  // 3) QKV GEMM: [4096,768] x [768,2304] -> split heads bf16
  gemm_bf16_wmma<0><<<dim3(3 * D_MODEL / BN, ROWS / BM), 256, 0, stream>>>(
      xn1, wqkv_b, ROWS, 3 * D_MODEL, D_MODEL, nullptr, nullptr, nullptr, Qb, Kb, Vb);

  // 4) flash attention per (b,h)
  flash_attn_wmma<<<dim3(SEQ / 64, BATCH * NHEADS), 128, 0, stream>>>(Qb, Kb, Vb, ctx);

  // 5) proj GEMM + residual: x1 = x + ctx*W_proj + b_proj
  gemm_bf16_wmma<1><<<dim3(D_MODEL / BN, ROWS / BM), 256, 0, stream>>>(
      ctx, wproj_b, ROWS, D_MODEL, D_MODEL, b_proj, x, x1, nullptr, nullptr, nullptr);

  // 6) LN2
  layernorm_bf16_kernel<<<ROWS, 256, 0, stream>>>(x1, ln2_g, ln2_b, xn2);

  // 7) FC1 + exact GELU -> bf16
  gemm_bf16_wmma<2><<<dim3(HIDDEN / BN, ROWS / BM), 256, 0, stream>>>(
      xn2, wfc1_b, ROWS, HIDDEN, D_MODEL, b_fc1, nullptr, nullptr, hbuf, nullptr, nullptr);

  // 8) FC2 + residual -> d_out fp32
  gemm_bf16_wmma<3><<<dim3(D_MODEL / BN, ROWS / BM), 256, 0, stream>>>(
      hbuf, wfc2_b, ROWS, D_MODEL, HIDDEN, b_fc2, x1, out, nullptr, nullptr, nullptr);
}
